// TransformerEncoderLayerAsGNN_9663676416887
// MI455X (gfx1250) — compile-verified, matrix operand order checked by reading, not run on hardware
//
#include <hip/hip_runtime.h>

// ---- problem constants (match reference) ----
#define B_   8
#define N_   1024
#define E_   512
#define H_   8
#define D_   64
#define FF_  2048
#define SCALE_ 0.125f        // D^-0.5 = 1/8
#define EPS_ 1e-5f

#define USE_ASYNC 1

// ---- CDNA5 WMMA types ----
typedef __attribute__((ext_vector_type(16))) __bf16 v16bf;
typedef __attribute__((ext_vector_type(8)))  float  v8f;

union Frag16 { v16bf v; unsigned short u[16]; };

__device__ __forceinline__ unsigned short f2bf(float f) {
  union { float f; unsigned u; } x; x.f = f;
  unsigned r = x.u + 0x7FFFu + ((x.u >> 16) & 1u);   // round-to-nearest-even
  return (unsigned short)(r >> 16);
}

__device__ __forceinline__ unsigned int pack_bf2(float lo, float hi) {
  return (unsigned int)f2bf(lo) | ((unsigned int)f2bf(hi) << 16);
}

__device__ __forceinline__ v8f wmma_bf16(const Frag16& a, const Frag16& b, v8f c) {
  return __builtin_amdgcn_wmma_f32_16x16x32_bf16(false, a.v, false, b.v,
                                                 (short)0, c, false, false);
}
// Same-B back-to-back WMMA: set reuse_b on the second instruction of the pair.
__device__ __forceinline__ v8f wmma_bf16_rb(const Frag16& a, const Frag16& b, v8f c) {
  return __builtin_amdgcn_wmma_f32_16x16x32_bf16(false, a.v, false, b.v,
                                                 (short)0, c, false, true);
}

// 16-byte global -> LDS copy via CDNA5 ASYNCcnt-tracked async load.
__device__ __forceinline__ void cp16_g2l(void* lds, const void* g) {
#if USE_ASYNC
  unsigned int       l = (unsigned int)(uintptr_t)lds;  // flat->LDS offset (addr[31:0])
  unsigned long long a = (unsigned long long)(uintptr_t)g;
  asm volatile("global_load_async_to_lds_b128 %0, %1, off"
               :: "v"(l), "v"(a) : "memory");
#else
  *(uint4*)lds = *(const uint4*)g;
#endif
}

template<int Ncnt>
__device__ __forceinline__ void wait_async_n() {
#if USE_ASYNC
  asm volatile("s_wait_asynccnt %0" :: "n"(Ncnt) : "memory");
#endif
}

// =====================================================================
// One-time converters
// =====================================================================
__global__ __launch_bounds__(256)
void cvt_bf16_kernel(const float* __restrict__ src, unsigned short* __restrict__ dst,
                     size_t n) {
  size_t i = ((size_t)blockIdx.x * 256 + threadIdx.x) * 8;
  if (i >= n) return;
  uint4 w;
  w.x = pack_bf2(src[i + 0], src[i + 1]);
  w.y = pack_bf2(src[i + 2], src[i + 3]);
  w.z = pack_bf2(src[i + 4], src[i + 5]);
  w.w = pack_bf2(src[i + 6], src[i + 7]);
  *(uint4*)(dst + i) = w;
}

// W[K,Nout] fp32 -> Wt[Nout,K] bf16 (one block per output row n)
__global__ __launch_bounds__(256)
void cvt_transpose_kernel(const float* __restrict__ W, unsigned short* __restrict__ Wt,
                          int K, int Nout) {
  const int n = blockIdx.x;
  unsigned int* dst = (unsigned int*)(Wt + (size_t)n * K);
  for (int k2 = threadIdx.x * 2; k2 < K; k2 += 512) {
    dst[k2 >> 1] = pack_bf2(W[(size_t)k2 * Nout + n], W[(size_t)(k2 + 1) * Nout + n]);
  }
}

// =====================================================================
// GEMM: C[M,Nout] = A[M,K](bf16) * Wt[Nout,K](bf16, pre-transposed) + bias
// OMODE 0: fp32 out.  1: bf16 row-major.  2: bf16 vT[b][h][d][n].
// 256 threads = 8 waves; 128x128 tile; wave = 32 rows x 64 cols
// (2 A-frags x 4 B-frags -> 8 WMMA per K-step); async double-buffered LDS.
// =====================================================================
#define LDA2_ 40   // padded K-stride in bf16 elements (80B, 16B-aligned)

template<int RELU, int OMODE>
__global__ __launch_bounds__(256)
void gemm_bf16_kernel(const unsigned short* __restrict__ A,
                      const unsigned short* __restrict__ Bt,
                      const float* __restrict__ bias, void* __restrict__ out,
                      int M, int K, int Nout)
{
  __shared__ unsigned short sA[2][128 * LDA2_];
  __shared__ unsigned short sB[2][128 * LDA2_];

  const int tid  = threadIdx.x;
  const int wave = tid >> 5;
  const int lane = tid & 31;
  const int half = lane >> 4;
  const int lidx = lane & 15;
  const int wr   = wave & 3;     // row group: 32 rows
  const int wc   = wave >> 2;    // col group: 64 cols
  const int row0 = blockIdx.y * 128;
  const int col0 = blockIdx.x * 128;

  v8f acc[2][4];
  v8f zero = {};
  #pragma unroll
  for (int rt = 0; rt < 2; ++rt)
    #pragma unroll
    for (int t = 0; t < 4; ++t) acc[rt][t] = zero;

  // stage one 128x32 A tile + 128x32 Bt tile (8KB each = 2x512 chunks)
  auto stage = [&](unsigned short* dA, unsigned short* dB, int k0) {
    #pragma unroll
    for (int i = 0; i < 2; ++i) {
      const int c = tid + i * 256;            // 0..511
      const int r = c >> 2, p = c & 3;
      cp16_g2l(dA + r * LDA2_ + p * 8, A  + (size_t)(row0 + r) * K + k0 + p * 8);
      cp16_g2l(dB + r * LDA2_ + p * 8, Bt + (size_t)(col0 + r) * K + k0 + p * 8);
    }
  };

  stage(sA[0], sB[0], 0);
  int cur = 0;
  for (int k0 = 0; k0 < K; k0 += 32) {
    if (k0 + 32 < K) {
      stage(sA[cur ^ 1], sB[cur ^ 1], k0 + 32);   // prefetch next tile
      wait_async_n<4>();                           // current tile landed
    } else {
      wait_async_n<0>();
    }
    __syncthreads();

    Frag16 a[2];
    #pragma unroll
    for (int rt = 0; rt < 2; ++rt) {
      const unsigned short* ap = sA[cur] + (wr * 32 + rt * 16 + lidx) * LDA2_ + half * 8;
      #pragma unroll
      for (int j = 0; j < 8; ++j) { a[rt].u[j] = ap[j]; a[rt].u[8 + j] = ap[16 + j]; }
    }
    #pragma unroll
    for (int t = 0; t < 4; ++t) {
      Frag16 b;
      const unsigned short* bp = sB[cur] + (wc * 64 + t * 16 + lidx) * LDA2_ + half * 8;
      #pragma unroll
      for (int j = 0; j < 8; ++j) { b.u[j] = bp[j]; b.u[8 + j] = bp[16 + j]; }
      acc[0][t] = wmma_bf16(a[0], b, acc[0][t]);
      acc[1][t] = wmma_bf16_rb(a[1], b, acc[1][t]);   // same B as previous WMMA
    }
    __syncthreads();
    cur ^= 1;
  }

  // epilogue: col = col0+64wc+16t+lidx ; row = row0+32wr+16rt+8half+m
  #pragma unroll
  for (int rt = 0; rt < 2; ++rt) {
    #pragma unroll
    for (int t = 0; t < 4; ++t) {
      const int col = col0 + wc * 64 + t * 16 + lidx;
      const float bv = bias[col];
      if (OMODE == 0) {
        float* C = (float*)out;
        #pragma unroll
        for (int m = 0; m < 8; ++m) {
          const int row = row0 + wr * 32 + rt * 16 + half * 8 + m;
          float v = acc[rt][t][m] + bv;
          if (RELU) v = fmaxf(v, 0.f);
          C[(size_t)row * Nout + col] = v;
        }
      } else if (OMODE == 1) {
        unsigned short* C = (unsigned short*)out;
        #pragma unroll
        for (int m = 0; m < 8; ++m) {
          const int row = row0 + wr * 32 + rt * 16 + half * 8 + m;
          float v = acc[rt][t][m] + bv;
          if (RELU) v = fmaxf(v, 0.f);
          C[(size_t)row * Nout + col] = f2bf(v);
        }
      } else {
        // vT[b][h][d][n]: rows m are consecutive n -> one 16B store
        unsigned short* vtp = (unsigned short*)out;
        const int hh = col >> 6, dd = col & 63;
        const int bb = row0 >> 10;   // 128-row tile stays within one b (1024%128==0)
        const int n0 = ((row0 + wr * 32 + rt * 16) & (N_ - 1)) + half * 8;
        uint4 w;
        w.x = pack_bf2(acc[rt][t][0] + bv, acc[rt][t][1] + bv);
        w.y = pack_bf2(acc[rt][t][2] + bv, acc[rt][t][3] + bv);
        w.z = pack_bf2(acc[rt][t][4] + bv, acc[rt][t][5] + bv);
        w.w = pack_bf2(acc[rt][t][6] + bv, acc[rt][t][7] + bv);
        *(uint4*)(vtp + ((size_t)(bb * H_ + hh) * D_ + dd) * N_ + n0) = w;
      }
    }
  }
}

// =====================================================================
// Flash attention: workgroup = (b,h) x 64-query block. 4 waves x 16 rows.
// K/V staging is async double-buffered.
// =====================================================================
#define LQ_ 72   // padded stride (144B, 16B-aligned)

__global__ __launch_bounds__(128)
void attention_kernel(const unsigned short* __restrict__ q,
                      const unsigned short* __restrict__ k,
                      const unsigned short* __restrict__ vt,
                      unsigned short* __restrict__ ctx)
{
  __shared__ unsigned short sQ[64 * LQ_];         // [qrow][d]
  __shared__ unsigned short sK[2][64 * LQ_];      // [key][d]
  __shared__ unsigned short sV[2][64 * LQ_];      // [d][key]
  __shared__ unsigned short sP[4][16 * LQ_];      // per-wave P tile [qrow][key]

  const int tid  = threadIdx.x;
  const int wave = tid >> 5;
  const int lane = tid & 31;
  const int half = lane >> 4;
  const int lidx = lane & 15;
  const int bh   = blockIdx.y;
  const int b    = bh / H_;
  const int h    = bh % H_;
  const int q0   = blockIdx.x * 64;

  auto stageKV = [&](int buf, int jb) {
    #pragma unroll
    for (int i = 0; i < 4; ++i) {
      const int c = tid + i * 128;
      const int r = c >> 3, p = c & 7;
      cp16_g2l(sK[buf] + r * LQ_ + p * 8,
               k + (size_t)(b * N_ + jb * 64 + r) * E_ + h * D_ + p * 8);
      cp16_g2l(sV[buf] + r * LQ_ + p * 8,
               vt + ((size_t)bh * D_ + r) * N_ + jb * 64 + p * 8);
    }
  };

  // stage Q block once
  #pragma unroll
  for (int i = 0; i < 4; ++i) {
    const int c = tid + i * 128;
    const int r = c >> 3, p = c & 7;
    cp16_g2l(sQ + r * LQ_ + p * 8, q + (size_t)(b * N_ + q0 + r) * E_ + h * D_ + p * 8);
  }
  wait_async_n<0>();
  __syncthreads();

  Frag16 aq[2];
  #pragma unroll
  for (int s = 0; s < 2; ++s) {
    const unsigned short* p = sQ + (wave * 16 + lidx) * LQ_ + s * 32 + half * 8;
    #pragma unroll
    for (int j = 0; j < 8; ++j) { aq[s].u[j] = p[j]; aq[s].u[8 + j] = p[16 + j]; }
  }

  v8f o[4];
  v8f zero = {};
  float rm[8], rl[8];
  #pragma unroll
  for (int t = 0; t < 4; ++t) o[t] = zero;
  #pragma unroll
  for (int m = 0; m < 8; ++m) { rm[m] = -1e30f; rl[m] = 0.f; }

  stageKV(0, 0);
  #pragma unroll 1
  for (int jb = 0; jb < N_ / 64; ++jb) {
    if (jb + 1 < N_ / 64) {
      stageKV((jb + 1) & 1, jb + 1);   // prefetch next K/V block
      wait_async_n<8>();
    } else {
      wait_async_n<0>();
    }
    __syncthreads();
    const unsigned short* cK = sK[jb & 1];
    const unsigned short* cV = sV[jb & 1];

    // S = Q K^T (contraction over d)
    v8f s[4];
    #pragma unroll
    for (int t = 0; t < 4; ++t) s[t] = zero;
    #pragma unroll
    for (int ks = 0; ks < 2; ++ks) {
      #pragma unroll
      for (int t = 0; t < 4; ++t) {
        Frag16 bk;
        const unsigned short* p = cK + (t * 16 + lidx) * LQ_ + ks * 32 + half * 8;
        #pragma unroll
        for (int j = 0; j < 8; ++j) { bk.u[j] = p[j]; bk.u[8 + j] = p[16 + j]; }
        s[t] = wmma_bf16(aq[ks], bk, s[t]);
      }
    }

    // online softmax; row m lives in VGPR m across a 16-lane half
    float bm[8];
    #pragma unroll
    for (int m = 0; m < 8; ++m) {
      float a0 = fmaxf(s[0][m], s[1][m]);
      float a1 = fmaxf(s[2][m], s[3][m]);
      bm[m] = fmaxf(a0, a1) * SCALE_;
    }
    #pragma unroll
    for (int mask = 1; mask < 16; mask <<= 1)
      #pragma unroll
      for (int m = 0; m < 8; ++m)
        bm[m] = fmaxf(bm[m], __shfl_xor(bm[m], mask, 32));

    float fac[8], rs[8];
    #pragma unroll
    for (int m = 0; m < 8; ++m) {
      float mn = fmaxf(rm[m], bm[m]);
      fac[m] = __expf(rm[m] - mn);
      rm[m] = mn;
      rs[m] = 0.f;
    }
    #pragma unroll
    for (int t = 0; t < 4; ++t)
      #pragma unroll
      for (int m = 0; m < 8; ++m) {
        float p = __expf(s[t][m] * SCALE_ - rm[m]);
        s[t][m] = p;
        rs[m] += p;
      }
    #pragma unroll
    for (int mask = 1; mask < 16; mask <<= 1)
      #pragma unroll
      for (int m = 0; m < 8; ++m)
        rs[m] += __shfl_xor(rs[m], mask, 32);
    #pragma unroll
    for (int m = 0; m < 8; ++m) rl[m] = rl[m] * fac[m] + rs[m];
    #pragma unroll
    for (int t = 0; t < 4; ++t)
      #pragma unroll
      for (int m = 0; m < 8; ++m) o[t][m] *= fac[m];

    // P (C-layout) -> wave-private LDS row-major; wave-local DS RAW is in-order
    #pragma unroll
    for (int t = 0; t < 4; ++t)
      #pragma unroll
      for (int m = 0; m < 8; ++m)
        sP[wave][(half * 8 + m) * LQ_ + t * 16 + lidx] = f2bf(s[t][m]);

    // O += P * V (contraction over key)
    #pragma unroll
    for (int ks = 0; ks < 2; ++ks) {
      Frag16 apf;
      const unsigned short* pp = sP[wave] + lidx * LQ_ + ks * 32 + half * 8;
      #pragma unroll
      for (int j = 0; j < 8; ++j) { apf.u[j] = pp[j]; apf.u[8 + j] = pp[16 + j]; }
      #pragma unroll
      for (int t = 0; t < 4; ++t) {
        Frag16 bv;
        const unsigned short* p = cV + (t * 16 + lidx) * LQ_ + ks * 32 + half * 8;
        #pragma unroll
        for (int j = 0; j < 8; ++j) { bv.u[j] = p[j]; bv.u[8 + j] = p[16 + j]; }
        o[t] = wmma_bf16(apf, bv, o[t]);
      }
    }
    __syncthreads();
  }

  // normalize + write ctx (bf16 row-major)
  #pragma unroll
  for (int t = 0; t < 4; ++t) {
    const int d = t * 16 + lidx;
    #pragma unroll
    for (int m = 0; m < 8; ++m) {
      const int row = q0 + wave * 16 + half * 8 + m;
      ctx[(size_t)(b * N_ + row) * E_ + h * D_ + d] = f2bf(o[t][m] / rl[m]);
    }
  }
}

// =====================================================================
// out = LayerNorm(a + r) * g + be ; optional secondary bf16 output.
// One wave per row (E=512 -> 16 per lane, pure shfl reductions).
// =====================================================================
__global__ __launch_bounds__(256)
void add_ln_kernel(const float* __restrict__ a, const float* __restrict__ r,
                   const float* __restrict__ g, const float* __restrict__ be,
                   float* __restrict__ out, unsigned short* __restrict__ out_bf,
                   int rows)
{
  const int wave = threadIdx.x >> 5;
  const int lane = threadIdx.x & 31;
  const int row  = blockIdx.x * 8 + wave;
  if (row >= rows) return;

  const float* pa = a + (size_t)row * E_;
  const float* pr = r + (size_t)row * E_;
  float vals[16];
  float s = 0.f;
  #pragma unroll
  for (int i = 0; i < 16; ++i) {
    float t = pa[lane + 32 * i] + pr[lane + 32 * i];
    vals[i] = t;
    s += t;
  }
  #pragma unroll
  for (int mask = 16; mask; mask >>= 1) s += __shfl_xor(s, mask, 32);
  const float mu = s * (1.f / E_);

  float vs = 0.f;
  #pragma unroll
  for (int i = 0; i < 16; ++i) { float d = vals[i] - mu; vs += d * d; }
  #pragma unroll
  for (int mask = 16; mask; mask >>= 1) vs += __shfl_xor(vs, mask, 32);
  const float inv = rsqrtf(vs * (1.f / E_) + EPS_);

  float* po = out + (size_t)row * E_;
  unsigned short* pb = out_bf ? out_bf + (size_t)row * E_ : nullptr;
  #pragma unroll
  for (int i = 0; i < 16; ++i) {
    const int c = lane + 32 * i;
    float v = (vals[i] - mu) * inv * g[c] + be[c];
    po[c] = v;
    if (pb) pb[c] = f2bf(v);
  }
}

// =====================================================================
extern "C" void kernel_launch(void* const* d_in, const int* in_sizes, int n_in,
                              void* d_out, int out_size, void* d_ws, size_t ws_size,
                              hipStream_t stream) {
  (void)in_sizes; (void)n_in; (void)out_size; (void)ws_size;
  const float* x   = (const float*)d_in[0];
  const float* Wq  = (const float*)d_in[1];
  const float* bq  = (const float*)d_in[2];
  const float* Wk  = (const float*)d_in[3];
  const float* bk  = (const float*)d_in[4];
  const float* Wv  = (const float*)d_in[5];
  const float* bv  = (const float*)d_in[6];
  const float* Wo  = (const float*)d_in[7];
  const float* bo  = (const float*)d_in[8];
  const float* g1  = (const float*)d_in[9];
  const float* b1  = (const float*)d_in[10];
  const float* g2  = (const float*)d_in[11];
  const float* b2  = (const float*)d_in[12];
  const float* W1  = (const float*)d_in[13];
  const float* bf1 = (const float*)d_in[14];
  const float* W2  = (const float*)d_in[15];
  const float* bf2 = (const float*)d_in[16];

  const int M = B_ * N_;                               // 8192
  char* wsb = (char*)d_ws;
  size_t cur = 0;
  auto take = [&](size_t bytes) { char* p = wsb + cur; cur += (bytes + 255) & ~(size_t)255; return p; };

  unsigned short* xbf  = (unsigned short*)take((size_t)M * E_ * 2);
  unsigned short* wqt  = (unsigned short*)take((size_t)E_ * E_ * 2);
  unsigned short* wkt  = (unsigned short*)take((size_t)E_ * E_ * 2);
  unsigned short* wvt  = (unsigned short*)take((size_t)E_ * E_ * 2);
  unsigned short* wot  = (unsigned short*)take((size_t)E_ * E_ * 2);
  unsigned short* w1t  = (unsigned short*)take((size_t)E_ * FF_ * 2);
  unsigned short* w2t  = (unsigned short*)take((size_t)E_ * FF_ * 2);
  // 32MB region: q/k/v/ctx (bf16), later reused for f1 (bf16, M x FF)
  char* big = take((size_t)M * FF_ * 2);
  unsigned short* qbf = (unsigned short*)big;
  unsigned short* kbf = (unsigned short*)(big + (size_t)M * E_ * 2);
  unsigned short* vtb = (unsigned short*)(big + (size_t)M * E_ * 2 * 2);
  unsigned short* ctb = (unsigned short*)(big + (size_t)M * E_ * 2 * 3);
  unsigned short* f1b = (unsigned short*)big;
  float* ao  = (float*)take((size_t)M * E_ * 4);       // reused for f2
  float* f2  = ao;
  float* h1  = (float*)take((size_t)M * E_ * 4);
  unsigned short* h1b = (unsigned short*)take((size_t)M * E_ * 2);

  dim3 blk(256);
  dim3 gE(E_ / 128, M / 128);                 // (4, 64)
  dim3 gF(FF_ / 128, M / 128);                // (16, 64)
  dim3 gAttn(N_ / 64, B_ * H_);               // (16, 64)
  dim3 gLN(M / 8);

  // one-time conversions
  cvt_bf16_kernel<<<dim3((M * E_) / (256 * 8)), 256, 0, stream>>>(x, xbf, (size_t)M * E_);
  cvt_transpose_kernel<<<dim3(E_),  256, 0, stream>>>(Wq, wqt, E_, E_);
  cvt_transpose_kernel<<<dim3(E_),  256, 0, stream>>>(Wk, wkt, E_, E_);
  cvt_transpose_kernel<<<dim3(E_),  256, 0, stream>>>(Wv, wvt, E_, E_);
  cvt_transpose_kernel<<<dim3(E_),  256, 0, stream>>>(Wo, wot, E_, E_);
  cvt_transpose_kernel<<<dim3(FF_), 256, 0, stream>>>(W1, w1t, E_, FF_);
  cvt_transpose_kernel<<<dim3(E_),  256, 0, stream>>>(W2, w2t, FF_, E_);

  // pipeline
  gemm_bf16_kernel<0, 1><<<gE, blk, 0, stream>>>(xbf, wqt, bq, qbf, M, E_, E_);
  gemm_bf16_kernel<0, 1><<<gE, blk, 0, stream>>>(xbf, wkt, bk, kbf, M, E_, E_);
  gemm_bf16_kernel<0, 2><<<gE, blk, 0, stream>>>(xbf, wvt, bv, vtb, M, E_, E_);
  attention_kernel<<<gAttn, dim3(128), 0, stream>>>(qbf, kbf, vtb, ctb);
  gemm_bf16_kernel<0, 0><<<gE, blk, 0, stream>>>(ctb, wot, bo, ao, M, E_, E_);
  add_ln_kernel<<<gLN, 256, 0, stream>>>(x, ao, g1, b1, h1, h1b, M);
  gemm_bf16_kernel<1, 1><<<gF, blk, 0, stream>>>(h1b, w1t, bf1, f1b, M, E_, FF_);
  gemm_bf16_kernel<0, 0><<<gE, blk, 0, stream>>>(f1b, w2t, bf2, f2, M, FF_, E_);
  add_ln_kernel<<<gLN, 256, 0, stream>>>(h1, f2, g2, b2, (float*)d_out, nullptr, M);
}
